// MultiHeadAttention_8950711846068
// MI455X (gfx1250) — matrix operand order repaired and checked by reading, not compiled, not run on hardware
//
#include <hip/hip_runtime.h>

// ---------------------------------------------------------------------------
// MHA for MI455X (gfx1250): f16 WMMA (v_wmma_f32_16x16x32_f16), wave32,
// async global->LDS double-buffered projection GEMMs (ASYNCcnt),
// 256 KB LDS score stripes per workgroup for fused softmax (320 KB/WGP).
// ---------------------------------------------------------------------------

typedef __attribute__((ext_vector_type(16))) _Float16 v16h;
typedef __attribute__((ext_vector_type(8)))  float    v8f;
typedef __attribute__((ext_vector_type(4)))  int      v4i;

union Frag16 { v16h v; _Float16 h[16]; unsigned u[8]; };

constexpr int Bc = 2, Sc = 2048, Dc = 1024, Hc = 16, DKc = 64;

__device__ __forceinline__ v8f wmma_f16(const Frag16& a, const Frag16& b, v8f c) {
  // 8 args: (neg_a, A, neg_b, B, c_mod, C, reuse_a, reuse_b)
  return __builtin_amdgcn_wmma_f32_16x16x32_f16(false, a.v, false, b.v,
                                                (short)0, c, false, false);
}

// ---- CDNA5 async global->LDS copy (16 B per lane), guarded ----------------
__device__ __forceinline__ void async_copy16(const _Float16* g, _Float16* l) {
#if __has_builtin(__builtin_amdgcn_global_load_async_to_lds_b128)
  __builtin_amdgcn_global_load_async_to_lds_b128(
      (__attribute__((address_space(1))) v4i*)g,
      (__attribute__((address_space(3))) v4i*)l, 0, 0);
#else
  *(uint4*)l = *(const uint4*)g;   // fallback: sync copy through VGPRs
#endif
}

__device__ __forceinline__ void async_wait0() {
#if __has_builtin(__builtin_amdgcn_s_wait_asynccnt)
  __builtin_amdgcn_s_wait_asynccnt(0);
#else
  asm volatile("s_wait_asynccnt 0x0" ::: "memory");
#endif
}

// ------------------------------ f32 -> f16 ---------------------------------
__global__ void cvt_f32_to_f16_kernel(const float* __restrict__ src,
                                      _Float16* __restrict__ dst, int n) {
  int i = blockIdx.x * 256 + threadIdx.x;
  if (i < n) dst[i] = (_Float16)src[i];
}

// ------------------------- projection GEMM ---------------------------------
// C[M=4096, N=1024] = A[f16, M x 1024] @ W^T   (W row-major [N,K], torch Linear)
// Block tile 128x64, K-step 32, async double-buffered LDS staging.
// mode 0: f32 out row-major (final projection into d_out)
// mode 1: f16 out, head-major [B,H,S,DK]          (Q, K)
// mode 2: f16 out, transposed  [B,H,DK,S]          (V)
__global__ void gemm_proj_kernel(const _Float16* __restrict__ A,
                                 const _Float16* __restrict__ W,
                                 void* __restrict__ out, int mode) {
  const int K = 1024;
  __shared__ _Float16 sA[2][128 * 32];  // 8 KB per stage
  __shared__ _Float16 sB[2][64 * 32];   // 4 KB per stage

  int tid = threadIdx.x;
  int lane = tid & 31;
  int w = tid >> 5;          // wave 0..7
  int waveM = w >> 1;        // 0..3
  int waveN = w & 1;         // 0..1
  int r  = lane & 15;
  int hi = lane >> 4;
  int m0 = blockIdx.x * 128;
  int n0 = blockIdx.y * 64;

  // Stage one 128x32 A tile + 64x32 B tile into LDS buffer `buf`.
  auto stage = [&](int buf, int kb) {
#pragma unroll
    for (int j = 0; j < 2; ++j) {               // A: 512 x 16B chunks
      int c = tid + 256 * j;
      int row = c >> 2, qq = c & 3;
      async_copy16(A + (size_t)(m0 + row) * K + kb + qq * 8,
                   &sA[buf][row * 32 + qq * 8]);
    }
    {                                            // B: 256 x 16B chunks
      int row = tid >> 2, qq = tid & 3;
      async_copy16(W + (size_t)(n0 + row) * K + kb + qq * 8,
                   &sB[buf][row * 32 + qq * 8]);
    }
  };

  stage(0, 0);
  v8f acc[2][2] = {};
  for (int it = 0; it < K / 32; ++it) {
    int buf = it & 1;
    async_wait0();          // my copies into `buf` are in LDS
    __syncthreads();        // everyone's copies landed; prev compute finished
    if (it + 1 < K / 32) stage(buf ^ 1, (it + 1) * 32);  // overlap next stage

    Frag16 a[2], b[2];
#pragma unroll
    for (int mt = 0; mt < 2; ++mt) {
      const _Float16* ap = &sA[buf][(waveM * 32 + mt * 16 + r) * 32];
#pragma unroll
      for (int i = 0; i < 8; ++i) {
        int k = (hi << 4) + (i << 1); // contiguous k = 16*hi + s
        a[mt].u[i] = *(const unsigned*)(ap + k);
      }
    }
#pragma unroll
    for (int nt = 0; nt < 2; ++nt) {
      const _Float16* bp = &sB[buf][(waveN * 32 + nt * 16 + r) * 32 + (hi << 4)];
#pragma unroll
      for (int i = 0; i < 8; ++i)
        b[nt].u[i] = *(const unsigned*)(bp + 2 * i);
    }
#pragma unroll
    for (int mt = 0; mt < 2; ++mt)
#pragma unroll
      for (int nt = 0; nt < 2; ++nt)
        acc[mt][nt] = wmma_f16(a[mt], b[nt], acc[mt][nt]);
  }

#pragma unroll
  for (int mt = 0; mt < 2; ++mt)
#pragma unroll
    for (int nt = 0; nt < 2; ++nt)
#pragma unroll
      for (int vr = 0; vr < 8; ++vr) {
        int m = m0 + waveM * 32 + mt * 16 + (hi << 3) + vr; // C/D: M = vr+hi*8
        int n = n0 + waveN * 32 + nt * 16 + r;
        float val = acc[mt][nt][vr];
        if (mode == 0) {
          ((float*)out)[(size_t)m * Dc + n] = val;
        } else {
          int bIdx = m >> 11;        // m / S
          int s    = m & 2047;
          int h    = n >> 6;         // n / DK
          int dk   = n & 63;
          if (mode == 1)
            ((_Float16*)out)[(((size_t)(bIdx * Hc + h) * Sc + s) << 6) + dk] =
                (_Float16)val;
          else
            ((_Float16*)out)[(((size_t)(bIdx * Hc + h) * DKc + dk) << 11) + s] =
                (_Float16)val;
        }
      }
}

// -------------------- fused scores + softmax + P@V -------------------------
// One block: 32 query rows x all 2048 keys, scores kept in 256 KB LDS.
__global__ void attention_kernel(const _Float16* __restrict__ Qh,
                                 const _Float16* __restrict__ Kh,
                                 const _Float16* __restrict__ Vt,
                                 const int* __restrict__ mask,
                                 float* __restrict__ attn_out,
                                 _Float16* __restrict__ ctx) {
  extern __shared__ float smem[];
  float* sc   = smem;              // [32][2048] scores / probs
  float* part = smem + 32 * 2048;  // [32][64] partial P@V

  int blk = blockIdx.x;
  int qt  = blk & 63;              // S/32 = 64 q-tiles
  int bh  = blk >> 6;              // b*H + h
  int b   = bh >> 4;
  int h   = bh & 15;
  int q0  = qt * 32;

  const _Float16* Q  = Qh + (size_t)bh * Sc * DKc;
  const _Float16* Kp = Kh + (size_t)bh * Sc * DKc;
  const _Float16* Vp = Vt + (size_t)bh * DKc * Sc;

  int tid = threadIdx.x, lane = tid & 31, w = tid >> 5;
  int r = lane & 15, hi = lane >> 4;

  // ---- Phase 1: scores = (Q K^T) / 8, masked, into LDS ----
  Frag16 aQ[2][2];
#pragma unroll
  for (int mt = 0; mt < 2; ++mt)
#pragma unroll
    for (int ks = 0; ks < 2; ++ks) {
      const _Float16* ap = Q + (size_t)(q0 + mt * 16 + r) * DKc + ks * 32;
#pragma unroll
      for (int i = 0; i < 8; ++i) {
        int k = (hi << 4) + (i << 1);
        aQ[mt][ks].u[i] = *(const unsigned*)(ap + k);
      }
    }

  for (int nt = w; nt < 128; nt += 8) {
    int n0 = nt * 16;
    Frag16 bK[2];
#pragma unroll
    for (int ks = 0; ks < 2; ++ks) {
      // B[k][n] = K[n][k]: contiguous along K rows
      const _Float16* bp = Kp + (size_t)(n0 + r) * DKc + ks * 32 + (hi << 4);
#pragma unroll
      for (int i = 0; i < 8; ++i)
        bK[ks].u[i] = *(const unsigned*)(bp + 2 * i);
    }
    v8f acc[2] = {};
#pragma unroll
    for (int mt = 0; mt < 2; ++mt)
#pragma unroll
      for (int ks = 0; ks < 2; ++ks)
        acc[mt] = wmma_f16(aQ[mt][ks], bK[ks], acc[mt]);

    int col  = n0 + r;
    int mval = mask[b * Sc + col];
#pragma unroll
    for (int mt = 0; mt < 2; ++mt)
#pragma unroll
      for (int vr = 0; vr < 8; ++vr) {
        int m = mt * 16 + (hi << 3) + vr;
        sc[m * 2048 + col] = mval ? acc[mt][vr] * 0.125f : -1e9f;
      }
  }
  __syncthreads();

  // ---- Phase 2: row softmax in LDS; write attn (f32) to HBM once ----
  for (int row = w; row < 32; row += 8) {
    float mx = -3.4e38f;
    for (int j = lane; j < 2048; j += 32) mx = fmaxf(mx, sc[row * 2048 + j]);
#pragma unroll
    for (int off = 16; off > 0; off >>= 1) mx = fmaxf(mx, __shfl_xor(mx, off, 32));
    float sum = 0.f;
    for (int j = lane; j < 2048; j += 32) {
      float e = __expf(sc[row * 2048 + j] - mx);
      sc[row * 2048 + j] = e;
      sum += e;
    }
#pragma unroll
    for (int off = 16; off > 0; off >>= 1) sum += __shfl_xor(sum, off, 32);
    float inv = 1.f / sum;
    float* arow = attn_out + ((size_t)bh * Sc + (q0 + row)) * Sc;
    for (int j = lane; j < 2048; j += 32) {
      float p = sc[row * 2048 + j] * inv;
      sc[row * 2048 + j] = p;
      arow[j] = p;                 // coalesced 128B/wave stores
    }
  }
  __syncthreads();

  // ---- Phase 3: ctx[32,64] = P[32,2048] @ V[2048,64] (split-K over waves) ----
  int ntile = w & 3;               // N: 4 tiles of 16
  int khalf = w >> 2;              // K halves: 0..1023 / 1024..2047
  v8f acc[2] = {};
  for (int ks = 0; ks < 32; ++ks) {
    int k0 = khalf * 1024 + ks * 32;
    Frag16 bV;
    {
      const _Float16* bp = Vp + (size_t)(ntile * 16 + r) * Sc + k0 + (hi << 4);
#pragma unroll
      for (int i = 0; i < 8; ++i)
        bV.u[i] = *(const unsigned*)(bp + 2 * i);
    }
#pragma unroll
    for (int mt = 0; mt < 2; ++mt) {
      Frag16 aP;
#pragma unroll
      for (int i = 0; i < 8; ++i) {
        int k = k0 + (hi << 4) + (i << 1);
        const float* sp = &sc[(mt * 16 + r) * 2048 + k];
        aP.h[2 * i]     = (_Float16)sp[0];
        aP.h[2 * i + 1] = (_Float16)sp[1];
      }
      acc[mt] = wmma_f16(aP, bV, acc[mt]);
    }
  }
  if (khalf == 1) {
#pragma unroll
    for (int mt = 0; mt < 2; ++mt)
#pragma unroll
      for (int vr = 0; vr < 8; ++vr) {
        int m = mt * 16 + (hi << 3) + vr;
        part[m * 64 + ntile * 16 + r] = acc[mt][vr];
      }
  }
  __syncthreads();
  if (khalf == 0) {
#pragma unroll
    for (int mt = 0; mt < 2; ++mt)
#pragma unroll
      for (int vr = 0; vr < 8; ++vr) {
        int m = mt * 16 + (hi << 3) + vr;
        int n = ntile * 16 + r;
        float vsum = acc[mt][vr] + part[m * 64 + n];
        int s = q0 + m;
        ctx[((size_t)(b * Sc + s)) * Dc + h * DKc + n] = (_Float16)vsum;
      }
  }
}

// ---------------------------------------------------------------------------
extern "C" void kernel_launch(void* const* d_in, const int* in_sizes, int n_in,
                              void* d_out, int out_size, void* d_ws, size_t ws_size,
                              hipStream_t stream) {
  (void)in_sizes; (void)n_in; (void)out_size; (void)ws_size;
  const float* q    = (const float*)d_in[0];
  const float* kin  = (const float*)d_in[1];
  const float* vin  = (const float*)d_in[2];
  const int*   mask = (const int*)  d_in[3];
  const float* Wq   = (const float*)d_in[4];
  const float* Wk   = (const float*)d_in[5];
  const float* Wv   = (const float*)d_in[6];
  const float* Wo   = (const float*)d_in[7];

  const size_t BSD = (size_t)Bc * Sc * Dc;   // 4,194,304
  const size_t DD  = (size_t)Dc * Dc;        // 1,048,576

  // workspace: 64 MB of f16
  _Float16* ws   = (_Float16*)d_ws;
  _Float16* Xq   = ws; ws += BSD;
  _Float16* Xk   = ws; ws += BSD;
  _Float16* Xv   = ws; ws += BSD;
  _Float16* Wq16 = ws; ws += DD;
  _Float16* Wk16 = ws; ws += DD;
  _Float16* Wv16 = ws; ws += DD;
  _Float16* Wo16 = ws; ws += DD;
  _Float16* Qh   = ws; ws += BSD;   // [B,H,S,DK]
  _Float16* Kh   = ws; ws += BSD;   // [B,H,S,DK]
  _Float16* Vt   = ws; ws += BSD;   // [B,H,DK,S]
  _Float16* Ctx  = ws;              // [B*S, D]

  float* out  = (float*)d_out;      // [B,S,D]
  float* attn = out + BSD;          // [B,H,S,S]

  auto cvt = [&](const float* s, _Float16* d, size_t n) {
    cvt_f32_to_f16_kernel<<<dim3((unsigned)((n + 255) / 256)), dim3(256), 0,
                            stream>>>(s, d, (int)n);
  };
  cvt(q,  Xq, BSD); cvt(kin, Xk, BSD); cvt(vin, Xv, BSD);
  cvt(Wq, Wq16, DD); cvt(Wk, Wk16, DD); cvt(Wv, Wv16, DD); cvt(Wo, Wo16, DD);

  dim3 ggrid(32, 16), gblk(256);
  gemm_proj_kernel<<<ggrid, gblk, 0, stream>>>(Xq, Wq16, (void*)Qh, 1);
  gemm_proj_kernel<<<ggrid, gblk, 0, stream>>>(Xk, Wk16, (void*)Kh, 1);
  gemm_proj_kernel<<<ggrid, gblk, 0, stream>>>(Xv, Wv16, (void*)Vt, 2);

  const int ATTN_SMEM = (32 * 2048 + 32 * 64) * (int)sizeof(float); // 270336 B
  (void)hipFuncSetAttribute(reinterpret_cast<const void*>(attention_kernel),
                            hipFuncAttributeMaxDynamicSharedMemorySize,
                            ATTN_SMEM);
  attention_kernel<<<dim3(Bc * Hc * (Sc / 32)), dim3(256), ATTN_SMEM, stream>>>(
      Qh, Kh, Vt, mask, attn, Ctx);

  gemm_proj_kernel<<<ggrid, gblk, 0, stream>>>(Ctx, Wo16, d_out, 0);
}
